// RelPosAttention_13288628814026
// MI455X (gfx1250) — matrix operand order repaired and checked by reading, not compiled, not run on hardware
//
#include <hip/hip_runtime.h>

#define DIM    768
#define NHEAD  12
#define HD     64
#define SEQ    1024
#define BATCH  8
#define ROWS   (BATCH*SEQ)     /* 8192 */
#define QKV_N  (3*DIM)         /* 2304 */
#define ATT_SCALE 0.125f       /* 64^-0.5 */

typedef __attribute__((ext_vector_type(16))) __bf16 v16bf;
typedef __attribute__((ext_vector_type(8)))  float  v8f;
typedef __attribute__((ext_vector_type(4)))  unsigned int v4u;
typedef __attribute__((ext_vector_type(8)))  int    v8i;
typedef __attribute__((ext_vector_type(4)))  int    v4i;

union BFrag { v16bf v; uint4 q[2]; };
union F8    { v8f v; float f[8]; };

__device__ __forceinline__ unsigned short f2bf(float x) {
  union { float f; unsigned u; } c; c.f = x;
  unsigned r = c.u + 0x7FFFu + ((c.u >> 16) & 1u);   // round-to-nearest-even
  return (unsigned short)(r >> 16);
}
__device__ __forceinline__ uint4 ld16(const unsigned short* p) {
  return *reinterpret_cast<const uint4*>(p);
}

// ---- Tensor Data Mover: DMA a 64-row x 32-col bf16 tile (row stride = strideElems)
// from global into LDS (contiguous, row-major). Issued once per wave; tracked by
// TENSORcnt. D# layout per CDNA5 ISA. This toolchain uses
// the 6-arg builtin: (g0 v4u, g1 v8i, g2 v4i, g3 v4i, v8i, cpol).
__device__ __forceinline__ void tdm_load_tile64x32(const unsigned short* gsrc,
                                                   unsigned ldsByteOff,
                                                   unsigned strideElems) {
  unsigned long long ga = (unsigned long long)(size_t)gsrc;
  v4u g0;
  g0.x = 1u;                                           // count=1, user mode
  g0.y = ldsByteOff;                                   // lds_addr (bytes)
  g0.z = (unsigned)(ga & 0xFFFFFFFFu);                 // global_addr[31:0]
  g0.w = (unsigned)((ga >> 32) & 0x01FFFFFFu) | (2u << 30);  // addr[56:32] | type=2
  v8i g1;
  g1[0] = 0x10000;                                     // data_size=1 (2 bytes)
  g1[1] = (int)((strideElems & 0xFFFFu) << 16);        // tensor_dim0[15:0] in [63:48]
  g1[2] = (int)(4096u << 16);                          // dim0[31:16]=0 | tensor_dim1=4096
  g1[3] = (int)(32u << 16);                            // dim1[31:16]=0 | tile_dim0=32
  g1[4] = 64;                                          // tile_dim1=64 | tile_dim2=0
  g1[5] = (int)strideElems;                            // tensor_dim0_stride[31:0]
  g1[6] = 0;                                           // stride hi | dim1_stride lo
  g1[7] = 0;
  v4i g2 = {0, 0, 0, 0};                               // 2D: dims 2/3 unused
  v4i g3 = {0, 0, 0, 0};
  v8i g4 = {0, 0, 0, 0, 0, 0, 0, 0};                   // extra group (unused, zero)
  __builtin_amdgcn_tensor_load_to_lds(g0, g1, g2, g3, g4, 0);
}

// ---------------- fp32 -> bf16 conversions ----------------
__global__ void cvt_bf16_kernel(const float* __restrict__ in,
                                unsigned short* __restrict__ out, int n) {
  int i = blockIdx.x * 256 + threadIdx.x;
  if (i < n) out[i] = f2bf(in[i]);
}

// in: rows x cols (row-major) -> out: cols x rows (row-major), bf16
__global__ void transpose_bf16_kernel(const float* __restrict__ in,
                                      unsigned short* __restrict__ out,
                                      int rows, int cols) {
  int i = blockIdx.x * 256 + threadIdx.x;
  if (i < rows * cols) {
    int r = i / cols, c = i - r * cols;
    out[(size_t)c * rows + r] = f2bf(in[i]);
  }
}

// ---------------- GEMM mainloop (shared): 64x64 C tile per 128-thr block ----------
// B^T tiles staged by the Tensor Data Mover into double-buffered LDS; A fragments
// loaded straight from global with the ISA 16-bit A layout. 4 back-to-back WMMAs
// per k-step, one barrier per iteration, DMA overlapped with math.
__device__ __forceinline__ void gemm_mainloop(const unsigned short* __restrict__ A,
                                              const unsigned short* __restrict__ Wt,
                                              unsigned short* lds,     // [2][64*32]
                                              int m0, int n0, int K,
                                              int wave, int g, int ln,
                                              F8 acc[4]) {
  const int rowA = m0 + wave * 16 + ln;
  const unsigned ldsBase = (unsigned)(size_t)lds;      // LDS aperture: addr[31:0]=offset

  if (wave == 0) {   // prologue: DMA tile k0=0 into buffer 0
    tdm_load_tile64x32(Wt + (size_t)n0 * K, ldsBase, (unsigned)K);
    __builtin_amdgcn_s_wait_tensorcnt(0);
  }
  __syncthreads();

  for (int k0 = 0; k0 < K; k0 += 32) {
    const int buf = (k0 >> 5) & 1;
    unsigned short* cur = lds + buf * (64 * 32);
    if (wave == 0 && (k0 + 32) < K) {   // DMA next tile into the other buffer
      tdm_load_tile64x32(Wt + (size_t)n0 * K + (k0 + 32),
                         ldsBase + (unsigned)((buf ^ 1) * (64 * 32 * 2)),
                         (unsigned)K);
    }

    BFrag a;   // A fragment: lane=row, K interleaved per ISA layout
    a.q[0] = ld16(A + (size_t)rowA * K + k0 + g * 8);
    a.q[1] = ld16(A + (size_t)rowA * K + k0 + 16 + g * 8);

    BFrag bf[4];   // preload all B fragments, then issue WMMAs back-to-back
#pragma unroll
    for (int tn = 0; tn < 4; ++tn) {
      const unsigned short* bp = cur + (tn * 16 + ln) * 32 + g * 8;
      bf[tn].q[0] = ld16(bp);
      bf[tn].q[1] = ld16(bp + 16);
    }
#pragma unroll
    for (int tn = 0; tn < 4; ++tn)
      acc[tn].v = __builtin_amdgcn_wmma_f32_16x16x32_bf16(
          false, a.v, false, bf[tn].v, (short)0, acc[tn].v, false, false);

    if (wave == 0) __builtin_amdgcn_s_wait_tensorcnt(0);
    __syncthreads();
  }
}

// ---------------- QKV GEMM: xb(8192x768) @ Wt^T(2304x768) + b ----------------
__global__ void qkv_gemm_kernel(const unsigned short* __restrict__ A,
                                const unsigned short* __restrict__ Wt,
                                const float* __restrict__ bias,
                                unsigned short* __restrict__ Qb,
                                unsigned short* __restrict__ Kb,
                                unsigned short* __restrict__ Vt) {
  __shared__ __align__(16) unsigned short ldsB[2][64 * 32];
  const int lane = threadIdx.x & 31;
  const int wave = threadIdx.x >> 5;
  const int g  = lane >> 4;
  const int ln = lane & 15;
  const int m0 = blockIdx.x * 64;
  const int n0 = blockIdx.y * 64;

  F8 acc[4];
#pragma unroll
  for (int t = 0; t < 4; ++t)
#pragma unroll
    for (int v = 0; v < 8; ++v) acc[t].f[v] = 0.f;

  gemm_mainloop(A, Wt, &ldsB[0][0], m0, n0, DIM, wave, g, ln, acc);

#pragma unroll
  for (int tn = 0; tn < 4; ++tn)
#pragma unroll
    for (int v = 0; v < 8; ++v) {
      int c   = n0 + tn * 16 + ln;
      int row = m0 + wave * 16 + v + 8 * g;
      float val = acc[tn].f[v] + bias[c];
      int which = c / DIM;             // uniform per block (DIM % 64 == 0)
      int rem   = c - which * DIM;
      int h = rem >> 6, d = rem & 63;
      int b = row >> 10, q = row & 1023;
      size_t bh = (size_t)b * NHEAD + h;
      if (which == 0)      Qb[(bh * SEQ + q) * HD + d] = f2bf(val * ATT_SCALE);
      else if (which == 1) Kb[(bh * SEQ + q) * HD + d] = f2bf(val);
      else                 Vt[(bh * HD + d) * SEQ + q] = f2bf(val);
    }
}

// ---------------- proj GEMM: ctx(8192x768) @ Wt^T(768x768) + b -> f32 -------------
__global__ void proj_gemm_kernel(const unsigned short* __restrict__ A,
                                 const unsigned short* __restrict__ Wt,
                                 const float* __restrict__ bias,
                                 float* __restrict__ out) {
  __shared__ __align__(16) unsigned short ldsB[2][64 * 32];
  const int lane = threadIdx.x & 31;
  const int wave = threadIdx.x >> 5;
  const int g  = lane >> 4;
  const int ln = lane & 15;
  const int m0 = blockIdx.x * 64;
  const int n0 = blockIdx.y * 64;

  F8 acc[4];
#pragma unroll
  for (int t = 0; t < 4; ++t)
#pragma unroll
    for (int v = 0; v < 8; ++v) acc[t].f[v] = 0.f;

  gemm_mainloop(A, Wt, &ldsB[0][0], m0, n0, DIM, wave, g, ln, acc);

#pragma unroll
  for (int tn = 0; tn < 4; ++tn)
#pragma unroll
    for (int v = 0; v < 8; ++v) {
      int c   = n0 + tn * 16 + ln;
      int row = m0 + wave * 16 + v + 8 * g;
      out[(size_t)row * DIM + c] = acc[tn].f[v] + bias[c];
    }
}

// ---------------- flash attention with rel-pos bias ----------------
// grid(96, 16), block 128: wave w handles q rows [(by*4+w)*16, +16)
__global__ void attn_kernel(const unsigned short* __restrict__ Qb,
                            const unsigned short* __restrict__ Kb,
                            const unsigned short* __restrict__ Vt,
                            const float* __restrict__ rel,
                            unsigned short* __restrict__ ctx) {
  __shared__ __align__(16) unsigned short ldsP[4][16 * 32];
  const int lane = threadIdx.x & 31;
  const int wave = threadIdx.x >> 5;
  const int g  = lane >> 4;
  const int ln = lane & 15;
  const int bh = blockIdx.x;
  const int b  = bh / NHEAD, h = bh - b * NHEAD;
  const int q0 = (blockIdx.y * 4 + wave) * 16;

  const unsigned short* Qp = Qb + (size_t)bh * SEQ * HD;
  const unsigned short* Kp = Kb + (size_t)bh * SEQ * HD;
  const unsigned short* Vp = Vt + (size_t)bh * HD * SEQ;
  const float* relh = rel + (size_t)h * SEQ * SEQ;
  unsigned short* myP = ldsP[wave];

  // Q fragments (16x64 = 2 chunks of 16x32), kept in registers
  BFrag qf[2];
  const int rowq = q0 + ln;
#pragma unroll
  for (int c = 0; c < 2; ++c) {
    qf[c].q[0] = ld16(Qp + (size_t)rowq * HD + c * 32 + g * 8);
    qf[c].q[1] = ld16(Qp + (size_t)rowq * HD + c * 32 + 16 + g * 8);
  }

  float m_i[8], l_i[8];
  F8 o[4];
#pragma unroll
  for (int v = 0; v < 8; ++v) { m_i[v] = -3.0e38f; l_i[v] = 0.f; }
#pragma unroll
  for (int t = 0; t < 4; ++t)
#pragma unroll
    for (int v = 0; v < 8; ++v) o[t].f[v] = 0.f;

  for (int j0 = 0; j0 < SEQ; j0 += 32) {
    __builtin_prefetch(relh + (size_t)(q0 + ln) * SEQ + j0 + 32, 0, 0);

    // S = Q @ K^T (+bias): two 16x16 score tiles
    F8 s[2];
#pragma unroll
    for (int t = 0; t < 2; ++t) {
#pragma unroll
      for (int v = 0; v < 8; ++v) s[t].f[v] = 0.f;
      const int key = j0 + t * 16 + ln;
#pragma unroll
      for (int c = 0; c < 2; ++c) {
        BFrag kf;   // B frag from K rows (N-major, k contiguous)
        kf.q[0] = ld16(Kp + (size_t)key * HD + c * 32 + g * 8);
        kf.q[1] = ld16(Kp + (size_t)key * HD + c * 32 + g * 8 + 16);
        s[t].v = __builtin_amdgcn_wmma_f32_16x16x32_bf16(
            false, qf[c].v, false, kf.v, (short)0, s[t].v, false, false);
      }
    }
    // + relative-position bias
#pragma unroll
    for (int t = 0; t < 2; ++t)
#pragma unroll
      for (int v = 0; v < 8; ++v)
        s[t].f[v] += relh[(size_t)(q0 + v + 8 * g) * SEQ + j0 + t * 16 + ln];

    // online softmax: row max across 16 lanes of this half
    float mt[8];
#pragma unroll
    for (int v = 0; v < 8; ++v) mt[v] = fmaxf(s[0].f[v], s[1].f[v]);
#pragma unroll
    for (int mask = 1; mask <= 8; mask <<= 1)
#pragma unroll
      for (int v = 0; v < 8; ++v)
        mt[v] = fmaxf(mt[v], __shfl_xor(mt[v], mask, 32));

    float p[2][8], rs[8];
#pragma unroll
    for (int v = 0; v < 8; ++v) {
      float mn = fmaxf(m_i[v], mt[v]);
      float corr = __expf(m_i[v] - mn);
      m_i[v] = mn;
      l_i[v] *= corr;
#pragma unroll
      for (int dt = 0; dt < 4; ++dt) o[dt].f[v] *= corr;
      float sum = 0.f;
#pragma unroll
      for (int t = 0; t < 2; ++t) { p[t][v] = __expf(s[t].f[v] - mn); sum += p[t][v]; }
      rs[v] = sum;
    }
#pragma unroll
    for (int mask = 1; mask <= 8; mask <<= 1)
#pragma unroll
      for (int v = 0; v < 8; ++v) rs[v] += __shfl_xor(rs[v], mask, 32);
#pragma unroll
    for (int v = 0; v < 8; ++v) l_i[v] += rs[v];

    // transpose P (C layout -> A layout) through per-wave LDS scratch
#pragma unroll
    for (int t = 0; t < 2; ++t)
#pragma unroll
      for (int v = 0; v < 8; ++v)
        myP[(v + 8 * g) * 32 + t * 16 + ln] = f2bf(p[t][v]);
    // same-wave DS ops are in-order; compiler inserts s_wait_dscnt before use
    BFrag pf;
    pf.q[0] = ld16(myP + ln * 32 + g * 8);
    pf.q[1] = ld16(myP + ln * 32 + 16 + g * 8);

    // O += P(16x32) @ V(32x64): V^T is n-major so B frags are contiguous
#pragma unroll
    for (int dt = 0; dt < 4; ++dt) {
      BFrag vf;
      const unsigned short* vp = Vp + (size_t)(dt * 16 + ln) * SEQ + j0 + g * 8;
      vf.q[0] = ld16(vp);
      vf.q[1] = ld16(vp + 16);
      o[dt].v = __builtin_amdgcn_wmma_f32_16x16x32_bf16(
          false, pf.v, false, vf.v, (short)0, o[dt].v, false, false);
    }
  }

  // normalize and write context (B, N, H*d) as bf16 for proj GEMM
#pragma unroll
  for (int dt = 0; dt < 4; ++dt)
#pragma unroll
    for (int v = 0; v < 8; ++v) {
      float val = o[dt].f[v] / l_i[v];
      int row = q0 + v + 8 * g;
      ctx[((size_t)b * SEQ + row) * DIM + h * HD + dt * 16 + ln] = f2bf(val);
    }
}

// ---------------- launcher ----------------
extern "C" void kernel_launch(void* const* d_in, const int* in_sizes, int n_in,
                              void* d_out, int out_size, void* d_ws, size_t ws_size,
                              hipStream_t stream) {
  const float* x      = (const float*)d_in[0];
  const float* relpos = (const float*)d_in[1];
  const float* qkv_w  = (const float*)d_in[2];
  const float* qkv_b  = (const float*)d_in[3];
  const float* proj_w = (const float*)d_in[4];
  const float* proj_b = (const float*)d_in[5];
  float* out = (float*)d_out;

  char* ws = (char*)d_ws;
  const size_t SZ_X    = (size_t)ROWS * DIM * 2;       // 12.6 MB
  const size_t SZ_WQ   = (size_t)QKV_N * DIM * 2;      // 3.5 MB
  const size_t SZ_WP   = (size_t)DIM * DIM * 2;        // 1.2 MB
  const size_t SZ_HEAD = (size_t)BATCH * NHEAD * SEQ * HD * 2;  // 12.6 MB

  unsigned short* xb    = (unsigned short*)(ws);
  unsigned short* wqkvt = (unsigned short*)(ws + SZ_X);
  unsigned short* wprjt = (unsigned short*)(ws + SZ_X + SZ_WQ);
  unsigned short* Qb    = (unsigned short*)(ws + SZ_X + SZ_WQ + SZ_WP);
  unsigned short* Kb    = (unsigned short*)(ws + SZ_X + SZ_WQ + SZ_WP + SZ_HEAD);
  unsigned short* Vt    = (unsigned short*)(ws + SZ_X + SZ_WQ + SZ_WP + 2 * SZ_HEAD);
  unsigned short* ctx   = (unsigned short*)(ws + SZ_X + SZ_WQ + SZ_WP + 3 * SZ_HEAD);

  const int nX = ROWS * DIM;
  cvt_bf16_kernel<<<(nX + 255) / 256, 256, 0, stream>>>(x, xb, nX);
  transpose_bf16_kernel<<<(DIM * QKV_N + 255) / 256, 256, 0, stream>>>(
      qkv_w, wqkvt, DIM, QKV_N);
  transpose_bf16_kernel<<<(DIM * DIM + 255) / 256, 256, 0, stream>>>(
      proj_w, wprjt, DIM, DIM);

  qkv_gemm_kernel<<<dim3(ROWS / 64, QKV_N / 64), 128, 0, stream>>>(
      xb, wqkvt, qkv_b, Qb, Kb, Vt);

  attn_kernel<<<dim3(BATCH * NHEAD, SEQ / 64), 128, 0, stream>>>(
      Qb, Kb, Vt, relpos, ctx);

  proj_gemm_kernel<<<dim3(ROWS / 64, DIM / 64), 128, 0, stream>>>(
      ctx, wprjt, proj_b, out);
}
